// LeafSelectionHead_49383533969729
// MI455X (gfx1250) — compile-verified, matrix operand order checked by reading, not run on hardware
//
#include <hip/hip_runtime.h>
#include <hip/hip_bf16.h>
#include <stdint.h>

typedef __attribute__((ext_vector_type(16))) _Float16 v16h;
typedef __attribute__((ext_vector_type(8)))  float    v8f;
typedef __attribute__((ext_vector_type(4)))  _Float16 v4h;

#define LEAKY_SLOPE 0.01f

// ---------------------------------------------------------------------------
// W1 [128,128] f32 row-major (k,n)  ->  W1^T [128,128] f16 row-major (n,k)
// ---------------------------------------------------------------------------
__global__ void w1t_kernel(const float* __restrict__ W1, _Float16* __restrict__ w1t) {
    int i = blockIdx.x * blockDim.x + threadIdx.x;   // 16384 threads
    int k = i >> 7, n = i & 127;
    w1t[n * 128 + k] = (_Float16)W1[k * 128 + n];
}

// ---------------------------------------------------------------------------
// deg[dst] += 1 over edges (reference: segment_sum(ones, dst))
// ---------------------------------------------------------------------------
__global__ void degree_kernel(const int* __restrict__ dst, float* __restrict__ deg, int n_edges) {
    int e = blockIdx.x * blockDim.x + threadIdx.x;
    if (e < n_edges) atomicAdd(&deg[dst[e]], 1.0f);
}

__global__ void dinv_kernel(const float* __restrict__ deg, float* __restrict__ dinv, int n_nodes) {
    int i = blockIdx.x * blockDim.x + threadIdx.x;
    if (i < n_nodes) dinv[i] = rsqrtf(deg[i] + 1.0f);
}

// ---------------------------------------------------------------------------
// hlin = x @ W1, stored f16.  One block = 16-row panel, 8 waves, wave w does
// output columns [16w, 16w+16).  K=128 in 4 WMMA steps of 32.
// ---------------------------------------------------------------------------
__global__ void gemm_kernel(const float* __restrict__ x, const _Float16* __restrict__ w1t,
                            _Float16* __restrict__ hlin, int n_nodes) {
    __shared__ __align__(16) _Float16 As[16 * 128];
    const int tid  = threadIdx.x;
    const int row0 = blockIdx.x * 16;

    for (int i = tid; i < 16 * 128; i += 256) {
        int r = i >> 7, c = i & 127;
        int gr = row0 + r;
        As[i] = (gr < n_nodes) ? (_Float16)x[(size_t)gr * 128 + c] : (_Float16)0.0f;
    }
    __syncthreads();

    const int wave = tid >> 5, lane = tid & 31;
    const int m    = lane & 15;
    const int kb   = (lane < 16) ? 0 : 8;   // K base for this half-wave
    const int ncol = wave * 16 + m;

    union Frag { v16h v; uint4 u[2]; } a, b;
    v8f acc = {};
#pragma unroll
    for (int k0 = 0; k0 < 128; k0 += 32) {
        // A 16x32 f16: elems 0-7 -> K=kb+e, elems 8-15 -> K=kb+8+e (two b128 LDS loads)
        a.u[0] = *(const uint4*)(&As[m * 128 + k0 + kb]);
        a.u[1] = *(const uint4*)(&As[m * 128 + k0 + kb + 16]);
        // B 32x16 f16 from W1^T (col-major of W1): same index pattern, N = ncol
        b.u[0] = *(const uint4*)(&w1t[(size_t)ncol * 128 + k0 + kb]);
        b.u[1] = *(const uint4*)(&w1t[(size_t)ncol * 128 + k0 + kb + 16]);
        acc = __builtin_amdgcn_wmma_f32_16x16x32_f16(false, a.v, false, b.v,
                                                     (short)0, acc, false, false);
    }

    const int mo = (lane < 16) ? 0 : 8;     // D: VGPR r -> row r (+8 upper half)
#pragma unroll
    for (int r = 0; r < 8; r++) {
        int row = row0 + r + mo;
        if (row < n_nodes) hlin[(size_t)row * 128 + ncol] = (_Float16)acc[r];
    }
}

// ---------------------------------------------------------------------------
// agg[i,:] = hlin[i,:] * dinv[i]^2 + b1   (self-loop + bias, also zero-inits agg)
// one thread per (node, 4 features)
// ---------------------------------------------------------------------------
__global__ void self_init_kernel(const _Float16* __restrict__ hlin, const float* __restrict__ dinv,
                                 const float* __restrict__ b1, float* __restrict__ agg, int n_nodes) {
    int t = blockIdx.x * blockDim.x + threadIdx.x;
    int i = t >> 5, q = (t & 31) * 4;
    if (i >= n_nodes) return;
    float di = dinv[i], s = di * di;
    v4h h4 = *(const v4h*)(&hlin[(size_t)i * 128 + q]);
    float4 bv = *(const float4*)(&b1[q]);
    float4 o;
    o.x = (float)h4[0] * s + bv.x;
    o.y = (float)h4[1] * s + bv.y;
    o.z = (float)h4[2] * s + bv.z;
    o.w = (float)h4[3] * s + bv.w;
    *(float4*)(&agg[(size_t)i * 128 + q]) = o;
}

// ---------------------------------------------------------------------------
// Edge aggregation: agg[dst,:] += norm * hlin[src,:].
// Each wave owns 32 edges per chunk: lanes load metadata for 32 edges, then
// the wave processes them one at a time via shuffle broadcast; per edge each
// lane moves 4 features (b64 f16 load + 4 f32 atomics).
// ---------------------------------------------------------------------------
__global__ void edge_agg_kernel(const int* __restrict__ src, const int* __restrict__ dst,
                                const _Float16* __restrict__ hlin, const float* __restrict__ dinv,
                                float* __restrict__ agg, int n_edges) {
    const int lane   = threadIdx.x & 31;
    const int gwave  = blockIdx.x * (blockDim.x >> 5) + (threadIdx.x >> 5);
    const int nwaves = gridDim.x * (blockDim.x >> 5);
    const int nchunk = (n_edges + 31) >> 5;
    const int q      = lane * 4;

    for (int c = gwave; c < nchunk; c += nwaves) {
        int e = (c << 5) + lane;
        int s = 0, d = 0;
        float w = 0.0f;
        if (e < n_edges) { s = src[e]; d = dst[e]; w = dinv[s] * dinv[d]; }  // w > 0 for real edges
#pragma unroll 4
        for (int i = 0; i < 32; i++) {
            float ww = __shfl(w, i);
            int   ss = __shfl(s, i);
            int   dd = __shfl(d, i);
            if (ww == 0.0f) continue;                 // only possible for tail padding
            v4h h4 = *(const v4h*)(&hlin[(size_t)ss * 128 + q]);
            float* ap = &agg[(size_t)dd * 128 + q];
            atomicAdd(ap + 0, ww * (float)h4[0]);
            atomicAdd(ap + 1, ww * (float)h4[1]);
            atomicAdd(ap + 2, ww * (float)h4[2]);
            atomicAdd(ap + 3, ww * (float)h4[3]);
        }
    }
}

// ---------------------------------------------------------------------------
// Per-node: h = leaky(agg); h2 = dot(h, W2); yleaf = h2*dinv^2 + b2;
// pooled[batch] += h; counts[batch] += 1.   One wave per node.
// ---------------------------------------------------------------------------
__global__ void node_head_kernel(const float* __restrict__ agg, const float* __restrict__ W2,
                                 const float* __restrict__ b2, const float* __restrict__ dinv,
                                 const int* __restrict__ batch,
                                 float* __restrict__ h2o, float* __restrict__ yleaf,
                                 float* __restrict__ pooled, float* __restrict__ counts,
                                 int n_nodes) {
    const int lane = threadIdx.x & 31;
    const int node = blockIdx.x * (blockDim.x >> 5) + (threadIdx.x >> 5);
    if (node >= n_nodes) return;
    const int q = lane * 4;

    float4 av = *(const float4*)(&agg[(size_t)node * 128 + q]);
    float h0 = av.x > 0.0f ? av.x : LEAKY_SLOPE * av.x;
    float h1 = av.y > 0.0f ? av.y : LEAKY_SLOPE * av.y;
    float hc = av.z > 0.0f ? av.z : LEAKY_SLOPE * av.z;
    float h3 = av.w > 0.0f ? av.w : LEAKY_SLOPE * av.w;

    float4 wv = *(const float4*)(&W2[q]);
    float p = h0 * wv.x + h1 * wv.y + hc * wv.z + h3 * wv.w;
#pragma unroll
    for (int off = 16; off > 0; off >>= 1) p += __shfl_xor(p, off);

    int b = batch[node];
    float* pp = &pooled[(size_t)b * 128 + q];
    atomicAdd(pp + 0, h0);
    atomicAdd(pp + 1, h1);
    atomicAdd(pp + 2, hc);
    atomicAdd(pp + 3, h3);

    if (lane == 0) {
        h2o[node] = p;
        float di = dinv[node];
        yleaf[node] = p * di * di + b2[0];
        atomicAdd(&counts[b], 1.0f);
    }
}

// yleaf[dst] += h2[src] * dinv[src] * dinv[dst]
__global__ void edge_leaf_kernel(const int* __restrict__ src, const int* __restrict__ dst,
                                 const float* __restrict__ h2, const float* __restrict__ dinv,
                                 float* __restrict__ yleaf, int n_edges) {
    int e = blockIdx.x * blockDim.x + threadIdx.x;
    if (e < n_edges) {
        int s = src[e], d = dst[e];
        atomicAdd(&yleaf[d], h2[s] * dinv[s] * dinv[d]);
    }
}

// yeos[g] = (pooled[g,:]/max(counts,1)) . W3 + b3
__global__ void eos_kernel(const float* __restrict__ pooled, const float* __restrict__ counts,
                           const float* __restrict__ W3, const float* __restrict__ b3,
                           float* __restrict__ yeos, int n_graphs) {
    int g = blockIdx.x * blockDim.x + threadIdx.x;
    if (g >= n_graphs) return;
    float inv = 1.0f / fmaxf(counts[g], 1.0f);
    float acc = 0.0f;
    for (int k = 0; k < 128; k++) acc += pooled[(size_t)g * 128 + k] * W3[k];
    yeos[g] = acc * inv + b3[0];
}

// ---------------------------------------------------------------------------
extern "C" void kernel_launch(void* const* d_in, const int* in_sizes, int n_in,
                              void* d_out, int out_size, void* d_ws, size_t ws_size,
                              hipStream_t stream) {
    const float* x     = (const float*)d_in[0];
    const int*   ei    = (const int*)d_in[1];
    const int*   batch = (const int*)d_in[2];
    const float* W1    = (const float*)d_in[3];
    const float* b1    = (const float*)d_in[4];
    const float* W2    = (const float*)d_in[5];
    const float* b2    = (const float*)d_in[6];
    const float* W3    = (const float*)d_in[7];
    const float* b3    = (const float*)d_in[8];

    const int n_nodes  = in_sizes[0] / 128;
    const int n_edges  = in_sizes[1] / 2;
    const int n_graphs = out_size - n_nodes;
    const int* src = ei;
    const int* dst = ei + n_edges;

    // workspace carve-out (256B aligned slices)
    char* ws = (char*)d_ws;
    size_t off = 0;
    auto carve = [&](size_t bytes) -> void* {
        void* p = ws + off;
        off = (off + bytes + 255) & ~(size_t)255;
        return p;
    };
    _Float16* hlin   = (_Float16*)carve((size_t)n_nodes * 128 * sizeof(_Float16));
    float*    agg    = (float*)   carve((size_t)n_nodes * 128 * sizeof(float));
    float*    deg    = (float*)   carve((size_t)n_nodes * sizeof(float));
    float*    dinv   = (float*)   carve((size_t)n_nodes * sizeof(float));
    float*    h2     = (float*)   carve((size_t)n_nodes * sizeof(float));
    _Float16* w1t    = (_Float16*)carve(128 * 128 * sizeof(_Float16));
    float*    pooled = (float*)   carve((size_t)n_graphs * 128 * sizeof(float));
    float*    counts = (float*)   carve((size_t)n_graphs * sizeof(float));

    float* yleaf = (float*)d_out;
    float* yeos  = (float*)d_out + n_nodes;

    // zero the accumulators that are atomically built each call
    hipMemsetAsync(deg,    0, (size_t)n_nodes * sizeof(float), stream);
    hipMemsetAsync(pooled, 0, (size_t)n_graphs * 128 * sizeof(float), stream);
    hipMemsetAsync(counts, 0, (size_t)n_graphs * sizeof(float), stream);

    w1t_kernel<<<(128 * 128) / 256, 256, 0, stream>>>(W1, w1t);
    degree_kernel<<<(n_edges + 255) / 256, 256, 0, stream>>>(dst, deg, n_edges);
    dinv_kernel<<<(n_nodes + 255) / 256, 256, 0, stream>>>(deg, dinv, n_nodes);
    gemm_kernel<<<(n_nodes + 15) / 16, 256, 0, stream>>>(x, w1t, hlin, n_nodes);
    self_init_kernel<<<(n_nodes * 32 + 255) / 256, 256, 0, stream>>>(hlin, dinv, b1, agg, n_nodes);
    edge_agg_kernel<<<2048, 256, 0, stream>>>(src, dst, hlin, dinv, agg, n_edges);
    node_head_kernel<<<(n_nodes + 7) / 8, 256, 0, stream>>>(agg, W2, b2, dinv, batch,
                                                            h2, yleaf, pooled, counts, n_nodes);
    edge_leaf_kernel<<<(n_edges + 255) / 256, 256, 0, stream>>>(src, dst, h2, dinv, yleaf, n_edges);
    eos_kernel<<<(n_graphs + 63) / 64, 64, 0, stream>>>(pooled, counts, W3, b3, yeos, n_graphs);
}
